// SubMLP_57526791963048
// MI455X (gfx1250) — matrix operand order repaired and checked by reading, not compiled, not run on hardware
//
#include <hip/hip_runtime.h>

// ---------------------------------------------------------------------------
// SubMLP on MI455X (gfx1250): BN-folded 4-GEMM pipeline, bf16 hi/lo split WMMA,
// double-buffered LDS with Tensor Data Mover (TDM) async tile loads.
// ---------------------------------------------------------------------------

typedef __attribute__((ext_vector_type(16))) __bf16 bf16x16;
typedef __attribute__((ext_vector_type(8)))  __bf16 bf16x8;
typedef __attribute__((ext_vector_type(4)))  __bf16 bf16x4;
typedef __attribute__((ext_vector_type(8)))  float  f32x8;
typedef __attribute__((ext_vector_type(4)))  unsigned int u32x4;
typedef __attribute__((ext_vector_type(8)))  int    i32x8;
typedef __attribute__((ext_vector_type(4)))  int    i32x4;

#define EPS 1e-5f

#if defined(__gfx1250__) && __has_builtin(__builtin_amdgcn_tensor_load_to_lds)
#define USE_TDM 1
#else
#define USE_TDM 0
#endif

#if __has_include(<hip/amd_detail/amd_gfx1250_TDM.h>)
#define TDM_SIX_ARG 1
#else
#define TDM_SIX_ARG 0
#endif

struct bfpair { __bf16 h, l; };

static __device__ __forceinline__ bf16x16 cat8(bf16x8 a, bf16x8 b) {
  return __builtin_shufflevector(a, b, 0,1,2,3,4,5,6,7,8,9,10,11,12,13,14,15);
}

static __device__ __forceinline__ f32x8 wmma_bf16(bf16x16 a, bf16x16 b, f32x8 c) {
  // D = A x B + C, 16x16x32 bf16 -> f32
  return __builtin_amdgcn_wmma_f32_16x16x32_bf16(false, a, false, b, (short)0, c,
                                                 false, false);
}

static __device__ __forceinline__ bfpair split_bf16(float x) {
  bfpair r;
  const __bf16 hh = (__bf16)x;
  r.h = hh;
  r.l = (__bf16)(x - (float)hh);
  return r;
}

#if USE_TDM
// Issue one TDM 2D tile load: 128 rows x 32 bf16 (64B/row), row stride = K
// elements, into LDS at byte offset lds_byte. Tensor dims == tile dims (all
// GEMM dims here are multiples of the tile, so no OOB handling needed).
static __device__ __forceinline__ void tdm_tile_load(const __bf16* g,
                                                     unsigned lds_byte, int K) {
  const unsigned long long ga = (unsigned long long)(uintptr_t)g;
  u32x4 g0;
  g0[0] = 1u;                                               // count=1 (valid)
  g0[1] = lds_byte;                                         // lds_addr
  g0[2] = (unsigned)ga;                                     // global_addr lo
  g0[3] = ((unsigned)(ga >> 32) & 0x01FFFFFFu) | 0x80000000u; // hi | type=2
  i32x8 g1;
  g1[0] = (int)(1u << 16);                    // data_size=2B, wg_mask=0
  g1[1] = (int)((32u & 0xFFFFu) << 16);       // tensor_dim0[15:0] (=32)
  g1[2] = (int)((32u >> 16) | ((128u & 0xFFFFu) << 16)); // td0 hi | td1 lo (=128)
  g1[3] = (int)((128u >> 16) | (32u << 16));  // td1 hi | tile_dim0=32
  g1[4] = (int)128;                            // tile_dim1=128, tile_dim2=0
  g1[5] = (int)(unsigned)K;                    // tensor_dim0_stride lo
  g1[6] = 0;                                   // stride0 hi | stride1 lo
  g1[7] = 0;
  const i32x4 z4 = {0, 0, 0, 0};
#if TDM_SIX_ARG
  const i32x8 z8 = {0, 0, 0, 0, 0, 0, 0, 0};
  __builtin_amdgcn_tensor_load_to_lds(g0, g1, z4, z4, z8, 0);
#else
  __builtin_amdgcn_tensor_load_to_lds(g0, g1, z4, z4, 0);
#endif
}
#endif

// ---------------------------------------------------------------------------
// GEMM: C[M,N] = A[M,K] @ W[N,K]^T + bias[N]
// A given as bf16 hi/lo pair, W as bf16 hi/lo pair (BN-folded effective W).
// l1mode==1: scatter columns into planar (p, b, o) layout (layer-1 output).
// Block: 256 threads = 8 waves (2x4); C tile 128x128; wave tile 64x32.
// LDS: double-buffered, 2 x (4 tiles x 8KB) = 64KB.
// ---------------------------------------------------------------------------
__global__ __launch_bounds__(256) void wmma_gemm(
    const __bf16* __restrict__ Ahi, const __bf16* __restrict__ Alo,
    const __bf16* __restrict__ Whi, const __bf16* __restrict__ Wlo,
    const float* __restrict__ bias, float* __restrict__ C,
    int M, int N, int K, int l1mode)
{
  __shared__ __align__(16) __bf16 smem[32768];   // 64 KB, two 32KB buffers

  const int tid  = threadIdx.x;
  const int lane = tid & 31;
  const int wave = tid >> 5;
  const int wr   = wave >> 2;       // 0..1  (M direction)
  const int wc   = wave & 3;        // 0..3  (N direction)
  const int hi   = (lane >> 4) & 1; // upper half-wave
  const int l16  = lane & 15;

  const size_t m0 = (size_t)blockIdx.y * 128;
  const size_t n0 = (size_t)blockIdx.x * 128;
  const int    nk = K >> 5;

  f32x8 acc[4][2] = {};

#if USE_TDM
  // TDM ignores EXEC: the issuing-wave guard MUST be a scalar (uniform)
  // branch, or every wave would duplicate the DMA. readfirstlane makes the
  // condition provably uniform -> s_cmp + s_cbranch, not exec masking.
  const bool w0 = (__builtin_amdgcn_readfirstlane(wave) == 0);
  const unsigned ldsB = (unsigned)(uintptr_t)(void*)&smem[0];
  if (w0) {   // preload buffer 0 (k0 = 0)
    tdm_tile_load(Ahi + m0 * K, ldsB + 0,     K);
    tdm_tile_load(Alo + m0 * K, ldsB + 8192,  K);
    tdm_tile_load(Whi + n0 * K, ldsB + 16384, K);
    tdm_tile_load(Wlo + n0 * K, ldsB + 24576, K);
  }
#else
  float4 ra[8];
  {
#pragma unroll
    for (int i = 0; i < 2; ++i) {
      const int c   = tid + i * 256;
      const int row = c >> 2;
      const int ce  = (c & 3) * 8;
      ra[0 + i] = *(const float4*)(Ahi + (m0 + row) * (size_t)K + ce);
      ra[2 + i] = *(const float4*)(Alo + (m0 + row) * (size_t)K + ce);
      ra[4 + i] = *(const float4*)(Whi + (n0 + row) * (size_t)K + ce);
      ra[6 + i] = *(const float4*)(Wlo + (n0 + row) * (size_t)K + ce);
    }
  }
#endif

  for (int ik = 0; ik < nk; ++ik) {
    __bf16* sb = smem + (ik & 1) * 16384;   // 32KB buffer stride (elements)

#if USE_TDM
    if (w0) __builtin_amdgcn_s_wait_tensorcnt(0);
    __syncthreads();                         // buf[ik] ready; buf[ik^1] free
    if (w0 && (ik + 1) < nk) {
      const unsigned nb = ldsB + (unsigned)(((ik + 1) & 1) * 32768);
      const size_t   kn = (size_t)(ik + 1) * 32;
      tdm_tile_load(Ahi + m0 * K + kn, nb + 0,     K);
      tdm_tile_load(Alo + m0 * K + kn, nb + 8192,  K);
      tdm_tile_load(Whi + n0 * K + kn, nb + 16384, K);
      tdm_tile_load(Wlo + n0 * K + kn, nb + 24576, K);
    }
#else
#pragma unroll
    for (int i = 0; i < 2; ++i) {
      const int c = tid + i * 256;
      ((float4*)(sb        ))[c] = ra[0 + i];
      ((float4*)(sb + 4096 ))[c] = ra[2 + i];
      ((float4*)(sb + 8192 ))[c] = ra[4 + i];
      ((float4*)(sb + 12288))[c] = ra[6 + i];
    }
    __syncthreads();
    if (ik + 1 < nk) {
      const size_t kn = (size_t)(ik + 1) * 32;
#pragma unroll
      for (int i = 0; i < 2; ++i) {
        const int c   = tid + i * 256;
        const int row = c >> 2;
        const int ce  = (c & 3) * 8;
        ra[0 + i] = *(const float4*)(Ahi + (m0 + row) * (size_t)K + kn + ce);
        ra[2 + i] = *(const float4*)(Alo + (m0 + row) * (size_t)K + kn + ce);
        ra[4 + i] = *(const float4*)(Whi + (n0 + row) * (size_t)K + kn + ce);
        ra[6 + i] = *(const float4*)(Wlo + (n0 + row) * (size_t)K + kn + ce);
      }
    }
#endif

    const __bf16* sAhi = sb;
    const __bf16* sAlo = sb + 4096;
    const __bf16* sWhi = sb + 8192;
    const __bf16* sWlo = sb + 12288;

    // B fragments (32x16): lanes 0-15 K in {0..7,16..23}; 16-31 in {8..15,24..31}.
    bf16x16 bh[2], bl[2];
#pragma unroll
    for (int ni = 0; ni < 2; ++ni) {
      const int cc = wc * 32 + ni * 16 + l16;
      const bf16x8* pb = (const bf16x8*)(sWhi + cc * 32);
      bh[ni] = cat8(pb[hi ? 1 : 0], pb[hi ? 3 : 2]);
      const bf16x8* pl = (const bf16x8*)(sWlo + cc * 32);
      bl[ni] = cat8(pl[hi ? 1 : 0], pl[hi ? 3 : 2]);
    }

#pragma unroll
    for (int mi = 0; mi < 4; ++mi) {
      // A fragments (16x32): lanes 0-15 K in {0..7,16..23}; 16-31 in {8..15,24..31}.
      const int r = wr * 64 + mi * 16 + l16;
      const bf16x8* pa = (const bf16x8*)(sAhi + r * 32);
      const bf16x16 ah = cat8(pa[hi ? 1 : 0], pa[hi ? 3 : 2]);
      const bf16x8* pal = (const bf16x8*)(sAlo + r * 32);
      const bf16x16 al = cat8(pal[hi ? 1 : 0], pal[hi ? 3 : 2]);
#pragma unroll
      for (int ni = 0; ni < 2; ++ni) {
        acc[mi][ni] = wmma_bf16(ah, bh[ni], acc[mi][ni]);  // hi*hi
        acc[mi][ni] = wmma_bf16(ah, bl[ni], acc[mi][ni]);  // hi*lo
        acc[mi][ni] = wmma_bf16(al, bh[ni], acc[mi][ni]);  // lo*hi
      }
    }
  }

  // Epilogue: C/D layout -> row = base + (hi?8:0) + vgpr, col = base_n + l16.
#pragma unroll
  for (int mi = 0; mi < 4; ++mi) {
#pragma unroll
    for (int ni = 0; ni < 2; ++ni) {
      const size_t col   = n0 + wc * 32 + ni * 16 + l16;
      const float  bv    = bias[col];
      const size_t rbase = m0 + wr * 64 + mi * 16 + hi * 8;
#pragma unroll
      for (int j = 0; j < 8; ++j) {
        const float  vv  = acc[mi][ni][j] + bv;
        const size_t row = rbase + j;
        if (l1mode) {
          const size_t p = col >> 10;
          const size_t o = col & 1023;
          C[p * (size_t)(8192 * 1024) + row * 1024 + o] = vv;
        } else {
          C[row * (size_t)N + col] = vv;
        }
      }
    }
  }
}

// ---------------------------------------------------------------------------
// Elementwise kernels
// ---------------------------------------------------------------------------

// Convert input x (8192x512 f32) into bf16 hi/lo pair (layer-1 A matrix).
__global__ void prep_in(const float* __restrict__ x,
                        __bf16* __restrict__ Ahi, __bf16* __restrict__ Alo) {
  const size_t base = ((size_t)blockIdx.x * blockDim.x + threadIdx.x) * 4;
  if (base >= (size_t)8192 * 512) return;
  const float4 v = *(const float4*)(x + base);
  bf16x4 h, l;
  bfpair p0 = split_bf16(v.x); h[0] = p0.h; l[0] = p0.l;
  bfpair p1 = split_bf16(v.y); h[1] = p1.h; l[1] = p1.l;
  bfpair p2 = split_bf16(v.z); h[2] = p2.h; l[2] = p2.l;
  bfpair p3 = split_bf16(v.w); h[3] = p3.h; l[3] = p3.l;
  *(bf16x4*)(Ahi + base) = h;
  *(bf16x4*)(Alo + base) = l;
}

// Effective weight: Weff[o][k] = sum_{p<nsum} a_p[o] * W[(p*CO+o)*CI + k],
// a_p = gamma * rsqrt(var + eps). nsum=1 with CO=3072 handles layer 1.
__global__ void prep_w(const float* __restrict__ W, const float* __restrict__ g,
                       const float* __restrict__ v,
                       __bf16* __restrict__ Whi, __bf16* __restrict__ Wlo,
                       int CO, int CI, int nsum) {
  const size_t t  = (size_t)blockIdx.x * blockDim.x + threadIdx.x;
  const int    kc = CI >> 2;
  if (t >= (size_t)CO * kc) return;
  const int o = (int)(t / kc);
  const int k = (int)(t % kc) * 4;
  float ax = 0.f, ay = 0.f, az = 0.f, aw = 0.f;
  for (int p = 0; p < nsum; ++p) {
    const int   go = p * CO + o;
    const float a  = g[go] * rsqrtf(v[go] + EPS);
    const float4 w = *(const float4*)(W + (size_t)go * CI + k);
    ax += a * w.x; ay += a * w.y; az += a * w.z; aw += a * w.w;
  }
  bf16x4 h, l;
  bfpair p0 = split_bf16(ax); h[0] = p0.h; l[0] = p0.l;
  bfpair p1 = split_bf16(ay); h[1] = p1.h; l[1] = p1.l;
  bfpair p2 = split_bf16(az); h[2] = p2.h; l[2] = p2.l;
  bfpair p3 = split_bf16(aw); h[3] = p3.h; l[3] = p3.l;
  *(bf16x4*)(Whi + (size_t)o * CI + k) = h;
  *(bf16x4*)(Wlo + (size_t)o * CI + k) = l;
}

// Effective bias: bias[o] = sum_p (beta - mean * a_p).
__global__ void prep_bias(const float* __restrict__ g, const float* __restrict__ be,
                          const float* __restrict__ me, const float* __restrict__ v,
                          float* __restrict__ bias, int CO, int nsum) {
  const int o = blockIdx.x * blockDim.x + threadIdx.x;
  if (o >= CO) return;
  float s = 0.f;
  for (int p = 0; p < nsum; ++p) {
    const int   go = p * CO + o;
    const float a  = g[go] * rsqrtf(v[go] + EPS);
    s += be[go] - me[go] * a;
  }
  bias[o] = s;
}

// Guided activation: mask = (sum over 3 planes > 0); write masked bf16 hi/lo.
__global__ void act_kernel(const float* __restrict__ C,
                           __bf16* __restrict__ Ahi, __bf16* __restrict__ Alo,
                           int H) {
  const size_t PS   = (size_t)8192 * H;
  const size_t base = ((size_t)blockIdx.x * blockDim.x + threadIdx.x) * 4;
  if (base >= PS) return;
  float4 c0 = *(const float4*)(C + base);
  float4 c1 = *(const float4*)(C + PS + base);
  float4 c2 = *(const float4*)(C + 2 * PS + base);
  const float mx = (c0.x + c1.x + c2.x) > 0.f ? 1.f : 0.f;
  const float my = (c0.y + c1.y + c2.y) > 0.f ? 1.f : 0.f;
  const float mz = (c0.z + c1.z + c2.z) > 0.f ? 1.f : 0.f;
  const float mw = (c0.w + c1.w + c2.w) > 0.f ? 1.f : 0.f;
  float4 cc[3] = {c0, c1, c2};
#pragma unroll
  for (int p = 0; p < 3; ++p) {
    bf16x4 h, l;
    bfpair q0 = split_bf16(cc[p].x * mx); h[0] = q0.h; l[0] = q0.l;
    bfpair q1 = split_bf16(cc[p].y * my); h[1] = q1.h; l[1] = q1.l;
    bfpair q2 = split_bf16(cc[p].z * mz); h[2] = q2.h; l[2] = q2.l;
    bfpair q3 = split_bf16(cc[p].w * mw); h[3] = q3.h; l[3] = q3.l;
    *(bf16x4*)(Ahi + (size_t)p * PS + base) = h;
    *(bf16x4*)(Alo + (size_t)p * PS + base) = l;
  }
}

// Final interleave: out[b, o, p] = C_plane[p][b][o], C planes are 8192x512.
__global__ void final_store(const float* __restrict__ C, float* __restrict__ out) {
  const size_t t = (size_t)blockIdx.x * blockDim.x + threadIdx.x;
  if (t >= (size_t)8192 * 512 * 3) return;
  const size_t p = t % 3;
  const size_t r = t / 3;
  const size_t o = r % 512;
  const size_t b = r / 512;
  out[t] = C[p * (size_t)(8192 * 512) + b * 512 + o];
}

// ---------------------------------------------------------------------------
// Workspace layout (bytes): ~392 MiB total
// ---------------------------------------------------------------------------
static constexpr size_t C_ELEMS  = (size_t)24576 * 2048;     // max f32 GEMM out
static constexpr size_t A_ELEMS  = (size_t)24576 * 2048;     // max bf16 A matrix
static constexpr size_t W_ELEMS  = (size_t)2048 * 1024;      // max effective W
static constexpr size_t AHI_OFF  = C_ELEMS * 4;
static constexpr size_t ALO_OFF  = AHI_OFF + A_ELEMS * 2;
static constexpr size_t WHI_OFF  = ALO_OFF + A_ELEMS * 2;
static constexpr size_t WLO_OFF  = WHI_OFF + W_ELEMS * 2;
static constexpr size_t BIAS_OFF = WLO_OFF + W_ELEMS * 2;

extern "C" void kernel_launch(void* const* d_in, const int* in_sizes, int n_in,
                              void* d_out, int out_size, void* d_ws, size_t ws_size,
                              hipStream_t stream) {
  (void)in_sizes; (void)n_in; (void)out_size; (void)ws_size;
  const float* x  = (const float*)d_in[0];
  const float* W1 = (const float*)d_in[1];
  const float* g1 = (const float*)d_in[2];
  const float* b1 = (const float*)d_in[3];
  const float* m1 = (const float*)d_in[4];
  const float* v1 = (const float*)d_in[5];
  const float* W2 = (const float*)d_in[6];
  const float* g2 = (const float*)d_in[7];
  const float* b2 = (const float*)d_in[8];
  const float* m2 = (const float*)d_in[9];
  const float* v2 = (const float*)d_in[10];
  const float* W3 = (const float*)d_in[11];
  const float* g3 = (const float*)d_in[12];
  const float* b3 = (const float*)d_in[13];
  const float* m3 = (const float*)d_in[14];
  const float* v3 = (const float*)d_in[15];
  const float* W4 = (const float*)d_in[16];
  const float* g4 = (const float*)d_in[17];
  const float* b4 = (const float*)d_in[18];
  const float* m4 = (const float*)d_in[19];
  const float* v4 = (const float*)d_in[20];
  float* out = (float*)d_out;

  char*   ws   = (char*)d_ws;
  float*  C    = (float*)(ws);
  __bf16* Ahi  = (__bf16*)(ws + AHI_OFF);
  __bf16* Alo  = (__bf16*)(ws + ALO_OFF);
  __bf16* Whi  = (__bf16*)(ws + WHI_OFF);
  __bf16* Wlo  = (__bf16*)(ws + WLO_OFF);
  float*  bias = (float*)(ws + BIAS_OFF);

  auto cdiv = [](size_t a, size_t b) { return (unsigned)((a + b - 1) / b); };

  // ---- Layer 1: 8192x3072x512, planar-scatter epilogue ----
  prep_in  <<<cdiv((size_t)8192 * 512 / 4, 256), 256, 0, stream>>>(x, Ahi, Alo);
  prep_w   <<<cdiv((size_t)3072 * 512 / 4, 256), 256, 0, stream>>>(W1, g1, v1, Whi, Wlo, 3072, 512, 1);
  prep_bias<<<cdiv(3072, 256), 256, 0, stream>>>(g1, b1, m1, v1, bias, 3072, 1);
  wmma_gemm<<<dim3(3072 / 128, 8192 / 128), 256, 0, stream>>>(Ahi, Alo, Whi, Wlo, bias, C,
                                                              8192, 3072, 512, 1);
  act_kernel<<<cdiv((size_t)8192 * 1024 / 4, 256), 256, 0, stream>>>(C, Ahi, Alo, 1024);

  // ---- Layer 2: 24576x2048x1024 ----
  prep_w   <<<cdiv((size_t)2048 * 1024 / 4, 256), 256, 0, stream>>>(W2, g2, v2, Whi, Wlo, 2048, 1024, 3);
  prep_bias<<<cdiv(2048, 256), 256, 0, stream>>>(g2, b2, m2, v2, bias, 2048, 3);
  wmma_gemm<<<dim3(2048 / 128, 24576 / 128), 256, 0, stream>>>(Ahi, Alo, Whi, Wlo, bias, C,
                                                               24576, 2048, 1024, 0);
  act_kernel<<<cdiv((size_t)8192 * 2048 / 4, 256), 256, 0, stream>>>(C, Ahi, Alo, 2048);

  // ---- Layer 3: 24576x1024x2048 ----
  prep_w   <<<cdiv((size_t)1024 * 2048 / 4, 256), 256, 0, stream>>>(W3, g3, v3, Whi, Wlo, 1024, 2048, 3);
  prep_bias<<<cdiv(1024, 256), 256, 0, stream>>>(g3, b3, m3, v3, bias, 1024, 3);
  wmma_gemm<<<dim3(1024 / 128, 24576 / 128), 256, 0, stream>>>(Ahi, Alo, Whi, Wlo, bias, C,
                                                               24576, 1024, 2048, 0);
  act_kernel<<<cdiv((size_t)8192 * 1024 / 4, 256), 256, 0, stream>>>(C, Ahi, Alo, 1024);

  // ---- Layer 4: 24576x512x1024 (no activation) ----
  prep_w   <<<cdiv((size_t)512 * 1024 / 4, 256), 256, 0, stream>>>(W4, g4, v4, Whi, Wlo, 512, 1024, 3);
  prep_bias<<<cdiv(512, 256), 256, 0, stream>>>(g4, b4, m4, v4, bias, 512, 3);
  wmma_gemm<<<dim3(512 / 128, 24576 / 128), 256, 0, stream>>>(Ahi, Alo, Whi, Wlo, bias, C,
                                                              24576, 512, 1024, 0);
  final_store<<<cdiv((size_t)8192 * 512 * 3, 256), 256, 0, stream>>>(C, out);
}
